// G2P_76802605187270
// MI455X (gfx1250) — compile-verified, matrix operand order checked by reading, not run on hardware
//
#include <hip/hip_runtime.h>
#include <math.h>

// ---------------------------------------------------------------------------
// G2P seq2seq (bi-GRU encoder + attention decoder) for MI455X / gfx1250.
// GEMMs: v_wmma_f32_16x16x32_bf16, async global->LDS tile staging (B128).
// ---------------------------------------------------------------------------

typedef __attribute__((ext_vector_type(16))) __bf16 v16bf;
typedef __attribute__((ext_vector_type(8)))  float  v8f;
typedef __attribute__((ext_vector_type(4)))  int    v4i;

#define DM      512     // D_MODEL
#define G3      1536    // 3*D
#define BATCH   256
#define SEQ_LT  128
#define SEQ_LP  64
#define D_PH    80
#define SCALE_ATT 0.044194173824159216f   // 1/sqrt(512)

__device__ __forceinline__ unsigned short f2bf(float f) {
    unsigned u = __float_as_uint(f);
    u += 0x7fffu + ((u >> 16) & 1u);       // round-to-nearest-even
    return (unsigned short)(u >> 16);
}

// ---- CDNA5 async global->LDS copy (ASYNCcnt-tracked), with fallback --------
#if defined(__has_builtin)
#if __has_builtin(__builtin_amdgcn_global_load_async_to_lds_b128) && \
    __has_builtin(__builtin_amdgcn_s_wait_asynccnt)
#define HAVE_ASYNC_LDS 1
#endif
#endif

#ifdef HAVE_ASYNC_LDS
typedef __attribute__((address_space(1))) v4i* gv4i_p;   // global (prints as __device__)
typedef __attribute__((address_space(3))) v4i* lv4i_p;   // LDS (prints as __shared__)
__device__ __forceinline__ void g2l_b128(const unsigned short* g, unsigned short* l) {
    __builtin_amdgcn_global_load_async_to_lds_b128(
        (gv4i_p)(uintptr_t)g, (lv4i_p)(uintptr_t)l, 0, 0);
}
__device__ __forceinline__ void async_join() { __builtin_amdgcn_s_wait_asynccnt(0); }
#else
__device__ __forceinline__ void g2l_b128(const unsigned short* g, unsigned short* l) {
    *(uint4*)l = *(const uint4*)g;          // global_load_b128 + ds_store_b128
}
__device__ __forceinline__ void async_join() {}
#endif

// ---------------------------------------------------------------------------
// Tiled GEMM:  C[M,N] = act( A[M,K](bf16) @ W[N,K]^T(bf16) + bias[N] )
// Workgroup tile 64x128, K-step 32, 8 waves: wave w -> rows 16*(w&3),
// col half 64*(w>>2); each wave owns four 16x16 WMMA tiles (A-frag reuse x4).
// LDS rows padded to 40 elems (80B) so B128 segments stay 16B aligned.
// ---------------------------------------------------------------------------
template<int ACT>   // 0 = none, 1 = tanh
__global__ __launch_bounds__(256)
void gemm_bf16_k(const unsigned short* __restrict__ A, int lda,
                 const unsigned short* __restrict__ W,
                 const float* __restrict__ bias,
                 float* __restrict__ Cf, int ldc,
                 unsigned short* __restrict__ Cb, int ldcb,
                 int M, int N, int K)
{
    __shared__ unsigned short As[64][40];
    __shared__ unsigned short Bs[128][40];
    const int tid  = threadIdx.x;
    const int lane = tid & 31, wave = tid >> 5;
    const int mt = wave & 3, nt = wave >> 2;
    const int m0 = blockIdx.y * 64, n0 = blockIdx.x * 128;

    v8f acc[4];
    #pragma unroll
    for (int s = 0; s < 4; ++s) acc[s] = (v8f){0.f,0.f,0.f,0.f,0.f,0.f,0.f,0.f};
    union Frag { v16bf v; unsigned u[8]; };

    const int arow = tid >> 2;                 // A: one B128 xfer per thread
    const int aseg = (tid & 3) << 3;           // element offset 0/8/16/24

    for (int k0 = 0; k0 < K; k0 += 32) {
        if (k0 + 32 < K)                       // speculative prefetch, next A tile
            __builtin_prefetch(A + (size_t)(m0 + arow) * lda + k0 + 32, 0, 0);

        // ---- stage A (64x32) tile ----
        g2l_b128(A + (size_t)(m0 + arow) * lda + k0 + aseg, &As[arow][aseg]);
        // ---- stage W (128x32) tile, OOB rows zero-filled ----
        #pragma unroll
        for (int i = 0; i < 2; ++i) {
            int idx = tid + (i << 8);
            int row = idx >> 2, seg = (idx & 3) << 3;
            if (n0 + row < N)
                g2l_b128(W + (size_t)(n0 + row) * K + k0 + seg, &Bs[row][seg]);
            else
                *(uint4*)&Bs[row][seg] = make_uint4(0u, 0u, 0u, 0u);
        }
        async_join();
        __syncthreads();

        // 16-bit operand lane pattern: lanes 0-15 -> K {0..7,16..23},
        // lanes 16-31 -> K {8..15,24..31}; B mirrors with N in place of M.
        const int mrow = (mt << 4) + (lane & 15);
        const int kb   = (lane >> 4) << 3;
        Frag fa;
        #pragma unroll
        for (int j = 0; j < 4; ++j) {
            fa.u[j]     = *(const unsigned*)&As[mrow][kb + 2*j];
            fa.u[4 + j] = *(const unsigned*)&As[mrow][kb + 16 + 2*j];
        }
        #pragma unroll
        for (int s = 0; s < 4; ++s) {
            const int nr = (nt << 6) + (s << 4) + (lane & 15);
            Frag fb;
            #pragma unroll
            for (int j = 0; j < 4; ++j) {
                fb.u[j]     = *(const unsigned*)&Bs[nr][kb + 2*j];
                fb.u[4 + j] = *(const unsigned*)&Bs[nr][kb + 16 + 2*j];
            }
            acc[s] = __builtin_amdgcn_wmma_f32_16x16x32_bf16(
                         false, fa.v, false, fb.v, (short)0, acc[s], false, false);
        }
        __syncthreads();
    }

    // C/D layout: VGPR r -> M = r + 8*(lane>>4), N = lane&15
    const int nlo = lane & 15;
    const int mhi = (lane >> 4) << 3;
    #pragma unroll
    for (int s = 0; s < 4; ++s) {
        int n = n0 + (nt << 6) + (s << 4) + nlo;
        if (n >= N) continue;
        float bv = bias ? bias[n] : 0.f;
        #pragma unroll
        for (int r = 0; r < 8; ++r) {
            int m = m0 + (mt << 4) + mhi + r;
            float x = acc[s][r] + bv;
            if (ACT == 1) x = tanhf(x);
            if (Cf) Cf[(size_t)m * ldc  + n] = x;
            if (Cb) Cb[(size_t)m * ldcb + n] = f2bf(x);
        }
    }
}

// ---------------------------------------------------------------------------
// GRU gate fusion: h' = (1-z)*n + z*h ; writes f32 state, bf16 state (for the
// next recurrent GEMM), and optionally the bf16 encoder output row.
// ---------------------------------------------------------------------------
__global__ __launch_bounds__(256)
void gru_pointwise_k(const float* __restrict__ gi, const float* __restrict__ gh,
                     float* __restrict__ h,
                     unsigned short* __restrict__ h_bf, int ld_hbf,
                     unsigned short* __restrict__ ys, int ys_stride)
{
    int b = blockIdx.x;
    #pragma unroll
    for (int i = 0; i < 2; ++i) {
        int d = threadIdx.x + (i << 8);
        size_t g = (size_t)b * G3 + d;
        float ir = gi[g], iz = gi[g + DM], inn = gi[g + 2*DM];
        float hr = gh[g], hz = gh[g + DM], hn  = gh[g + 2*DM];
        float r = 1.f / (1.f + __expf(-(ir + hr)));
        float z = 1.f / (1.f + __expf(-(iz + hz)));
        float n = tanhf(inn + r * hn);
        size_t ho = (size_t)b * DM + d;
        float hv = (1.f - z) * n + z * h[ho];
        h[ho] = hv;
        unsigned short bv = f2bf(hv);
        h_bf[(size_t)b * ld_hbf + d] = bv;
        if (ys) ys[(size_t)b * ys_stride + d] = bv;
    }
}

// ---------------------------------------------------------------------------
// Attention: scores = q.k * SCALE, softmax over LT, a_o = a @ v.
// One block (128 threads) per batch row; q staged in LDS; probs written
// directly into d_out attn slice; a_o written as bf16 into decoder GEMM input.
// ---------------------------------------------------------------------------
__global__ __launch_bounds__(128)
void attend_k(const float* __restrict__ q, const float* __restrict__ k,
              const float* __restrict__ v,
              float* __restrict__ attn_out,          // + b*(LP*LT) + l
              unsigned short* __restrict__ ao_bf)    // + b*1024 + d
{
    __shared__ float qs[DM];
    __shared__ float red[SEQ_LT];
    int b = blockIdx.x, l = threadIdx.x;
    #pragma unroll
    for (int i = 0; i < 4; ++i)
        qs[l + (i << 7)] = q[(size_t)b * DM + l + (i << 7)];
    __syncthreads();

    const float* kr = k + ((size_t)b * SEQ_LT + l) * DM;
    float dot = 0.f;
    for (int d = 0; d < DM; ++d) dot += qs[d] * kr[d];
    float s = dot * SCALE_ATT;

    red[l] = s; __syncthreads();
    for (int o = 64; o > 0; o >>= 1) { if (l < o) red[l] = fmaxf(red[l], red[l + o]); __syncthreads(); }
    float mx = red[0]; __syncthreads();
    float e = __expf(s - mx);
    red[l] = e; __syncthreads();
    for (int o = 64; o > 0; o >>= 1) { if (l < o) red[l] += red[l + o]; __syncthreads(); }
    float a = e / red[0];
    attn_out[(size_t)b * (SEQ_LP * SEQ_LT) + l] = a;
    __syncthreads();
    red[l] = a; __syncthreads();

    #pragma unroll
    for (int i = 0; i < 4; ++i) {
        int d = l + (i << 7);
        float acc = 0.f;
        for (int lt = 0; lt < SEQ_LT; ++lt)
            acc += red[lt] * v[((size_t)b * SEQ_LT + lt) * DM + d];
        ao_bf[(size_t)b * 1024 + d] = f2bf(acc);
    }
}

// ---------------------------------------------------------------------------
// Utilities: gathers, conversion, zero, argmax
// ---------------------------------------------------------------------------
__global__ __launch_bounds__(256)
void embed_text_k(const int* __restrict__ text, const float* __restrict__ Ea,
                  unsigned short* __restrict__ embT)   // time-major [LT,B,D]
{
    int row = blockIdx.x;                // t*B + b
    int t = row >> 8, b = row & 255;
    int tok = text[b * SEQ_LT + t];
    #pragma unroll
    for (int i = 0; i < 2; ++i) {
        int d = threadIdx.x + (i << 8);
        embT[(size_t)row * DM + d] = f2bf(Ea[(size_t)tok * DM + d]);
    }
}

__global__ __launch_bounds__(256)
void embed_ph_k(const int* __restrict__ dec_i, const float* __restrict__ Eph,
                unsigned short* __restrict__ xin)      // rows of 1024, d in [0,512)
{
    int b = blockIdx.x;
    int tok = dec_i[b];
    #pragma unroll
    for (int i = 0; i < 2; ++i) {
        int d = threadIdx.x + (i << 8);
        xin[(size_t)b * 1024 + d] = f2bf(Eph[(size_t)tok * DM + d]);
    }
}

__global__ void f32_to_bf16_k(const float* __restrict__ in,
                              unsigned short* __restrict__ out, int n)
{
    int i = blockIdx.x * blockDim.x + threadIdx.x;
    if (i < n) out[i] = f2bf(in[i]);
}

__global__ void zero_u32_k(unsigned* __restrict__ p, int n)
{
    int i = blockIdx.x * blockDim.x + threadIdx.x;
    if (i < n) p[i] = 0u;
}

__global__ void argmax_k(const float* __restrict__ o, int stride,
                         int* __restrict__ dec_i)
{
    int b = blockIdx.x * blockDim.x + threadIdx.x;
    if (b < BATCH) {
        const float* r = o + (size_t)b * stride;
        float best = r[0]; int bi = 0;
        for (int j = 1; j < D_PH; ++j) { float v = r[j]; if (v > best) { best = v; bi = j; } }
        dec_i[b] = bi;
    }
}

// ---------------------------------------------------------------------------
// Host orchestration
// ---------------------------------------------------------------------------
extern "C" void kernel_launch(void* const* d_in, const int* in_sizes, int n_in,
                              void* d_out, int out_size, void* d_ws, size_t ws_size,
                              hipStream_t stream)
{
    (void)in_sizes; (void)n_in; (void)out_size; (void)ws_size;

    const int*   text      = (const int*)  d_in[0];
    const float* E_alpha   = (const float*)d_in[2];
    const float* enc_Wih_f = (const float*)d_in[3];
    const float* enc_Whh_f = (const float*)d_in[4];
    const float* enc_bih_f = (const float*)d_in[5];
    const float* enc_bhh_f = (const float*)d_in[6];
    const float* enc_Wih_b = (const float*)d_in[7];
    const float* enc_Whh_b = (const float*)d_in[8];
    const float* enc_bih_b = (const float*)d_in[9];
    const float* enc_bhh_b = (const float*)d_in[10];
    const float* hpost_W   = (const float*)d_in[11];
    const float* hpost_b   = (const float*)d_in[12];
    const float* wk_W      = (const float*)d_in[13];
    const float* wk_b      = (const float*)d_in[14];
    const float* wv_W      = (const float*)d_in[15];
    const float* wv_b      = (const float*)d_in[16];
    const float* E_ph      = (const float*)d_in[17];
    const float* dec_Wih   = (const float*)d_in[18];
    const float* dec_Whh   = (const float*)d_in[19];
    const float* dec_bih   = (const float*)d_in[20];
    const float* dec_bhh   = (const float*)d_in[21];
    const float* post_W    = (const float*)d_in[22];
    const float* post_b    = (const float*)d_in[23];

    float* out_attn = (float*)d_out;                              // [B,LP,LT]
    float* out_res  = out_attn + (size_t)BATCH * SEQ_LP * SEQ_LT; // [B,LP,80]

    // ---- workspace carving ----
    char* wp = (char*)d_ws;
    auto alloc  = [&](size_t bytes) -> void* {
        void* r = (void*)wp; wp += (bytes + 255) & ~(size_t)255; return r;
    };
    auto allocb = [&](size_t e) { return (unsigned short*)alloc(e * 2); };
    auto allocf = [&](size_t e) { return (float*)alloc(e * 4); };

    unsigned short* wWihF  = allocb((size_t)G3 * DM);
    unsigned short* wWhhF  = allocb((size_t)G3 * DM);
    unsigned short* wWihB  = allocb((size_t)G3 * DM);
    unsigned short* wWhhB  = allocb((size_t)G3 * DM);
    unsigned short* wHpost = allocb((size_t)DM * 2 * DM);
    unsigned short* wWk    = allocb((size_t)DM * 2 * DM);
    unsigned short* wWv    = allocb((size_t)DM * 2 * DM);
    unsigned short* wDWih  = allocb((size_t)G3 * 2 * DM);
    unsigned short* wDWhh  = allocb((size_t)G3 * DM);
    unsigned short* wPost  = allocb((size_t)D_PH * DM);

    unsigned short* embT   = allocb((size_t)SEQ_LT * BATCH * DM);   // [LT,B,D]
    float* GIf             = allocf((size_t)SEQ_LT * BATCH * G3);   // x@Wih.T+b, precomputed
    float* GIb             = allocf((size_t)SEQ_LT * BATCH * G3);
    float* ghF             = allocf((size_t)BATCH * G3);
    float* ghB             = allocf((size_t)BATCH * G3);
    float* hF              = allocf((size_t)BATCH * DM);
    float* hB              = allocf((size_t)BATCH * DM);
    unsigned short* hcat   = allocb((size_t)BATCH * 2 * DM);        // [hF_bf | hB_bf]
    unsigned short* enc_o  = allocb((size_t)BATCH * SEQ_LT * 2 * DM);
    float* kbuf            = allocf((size_t)BATCH * SEQ_LT * DM);
    float* vbuf            = allocf((size_t)BATCH * SEQ_LT * DM);
    float* hdec            = allocf((size_t)BATCH * DM);
    unsigned short* hdecb  = allocb((size_t)BATCH * DM);
    unsigned short* xin    = allocb((size_t)BATCH * 2 * DM);        // [e | a_o] bf16
    float* gid             = allocf((size_t)BATCH * G3);
    float* ghd             = allocf((size_t)BATCH * G3);
    int*   dec_i           = (int*)alloc((size_t)BATCH * 4);

    // ---- helpers ----
    auto cvt = [&](const float* s, unsigned short* d, int n) {
        f32_to_bf16_k<<<(n + 255) / 256, 256, 0, stream>>>(s, d, n);
    };
    auto zero = [&](void* ptr, size_t words) {
        zero_u32_k<<<((int)words + 255) / 256, 256, 0, stream>>>((unsigned*)ptr, (int)words);
    };
    auto gemm = [&](const unsigned short* A, int lda, const unsigned short* W,
                    const float* bias, float* Cf, int ldc,
                    unsigned short* Cb, int ldcb, int M, int N, int K, int act) {
        dim3 g((N + 127) / 128, M / 64), blk(256);
        if (act) gemm_bf16_k<1><<<g, blk, 0, stream>>>(A, lda, W, bias, Cf, ldc, Cb, ldcb, M, N, K);
        else     gemm_bf16_k<0><<<g, blk, 0, stream>>>(A, lda, W, bias, Cf, ldc, Cb, ldcb, M, N, K);
    };

    // ---- weight conversion (idempotent) ----
    cvt(enc_Wih_f, wWihF,  G3 * DM);
    cvt(enc_Whh_f, wWhhF,  G3 * DM);
    cvt(enc_Wih_b, wWihB,  G3 * DM);
    cvt(enc_Whh_b, wWhhB,  G3 * DM);
    cvt(hpost_W,   wHpost, DM * 2 * DM);
    cvt(wk_W,      wWk,    DM * 2 * DM);
    cvt(wv_W,      wWv,    DM * 2 * DM);
    cvt(dec_Wih,   wDWih,  G3 * 2 * DM);
    cvt(dec_Whh,   wDWhh,  G3 * DM);
    cvt(post_W,    wPost,  D_PH * DM);

    // ---- state init ----
    zero(hF,   (size_t)BATCH * DM);
    zero(hB,   (size_t)BATCH * DM);
    zero(hcat, (size_t)BATCH * DM);   // bf16 words -> u32 count
    zero(dec_i, BATCH);

    // ---- encoder ----
    embed_text_k<<<SEQ_LT * BATCH, 256, 0, stream>>>(text, E_alpha, embT);

    // fully parallel precompute of all input projections (both directions)
    gemm(embT, DM, wWihF, enc_bih_f, GIf, G3, nullptr, 0, SEQ_LT * BATCH, G3, DM, 0);
    gemm(embT, DM, wWihB, enc_bih_b, GIb, G3, nullptr, 0, SEQ_LT * BATCH, G3, DM, 0);

    // sequential recurrences (forward & backward interleaved on stream)
    for (int t = 0; t < SEQ_LT; ++t) {
        gemm(hcat, 2 * DM, wWhhF, enc_bhh_f, ghF, G3, nullptr, 0, BATCH, G3, DM, 0);
        gru_pointwise_k<<<BATCH, 256, 0, stream>>>(
            GIf + (size_t)t * BATCH * G3, ghF, hF, hcat, 2 * DM,
            enc_o + (size_t)t * 2 * DM, SEQ_LT * 2 * DM);
        int tr = SEQ_LT - 1 - t;
        gemm(hcat + DM, 2 * DM, wWhhB, enc_bhh_b, ghB, G3, nullptr, 0, BATCH, G3, DM, 0);
        gru_pointwise_k<<<BATCH, 256, 0, stream>>>(
            GIb + (size_t)tr * BATCH * G3, ghB, hB, hcat + DM, 2 * DM,
            enc_o + (size_t)tr * 2 * DM + DM, SEQ_LT * 2 * DM);
    }

    // h = tanh([hF,hB] @ hpost_W.T + b)   (dual f32 + bf16 store)
    gemm(hcat, 2 * DM, wHpost, hpost_b, hdec, DM, hdecb, DM, BATCH, DM, 2 * DM, 1);

    // k, v projections over all encoder outputs
    gemm(enc_o, 2 * DM, wWk, wk_b, kbuf, DM, nullptr, 0, BATCH * SEQ_LT, DM, 2 * DM, 0);
    gemm(enc_o, 2 * DM, wWv, wv_b, vbuf, DM, nullptr, 0, BATCH * SEQ_LT, DM, 2 * DM, 0);

    // initial attention -> attn[:,0,:], a_o -> xin[:,512:]
    attend_k<<<BATCH, 128, 0, stream>>>(hdec, kbuf, vbuf, out_attn, xin + DM);

    // ---- decoder (64 greedy steps) ----
    for (int ti = 0; ti < SEQ_LP; ++ti) {
        embed_ph_k<<<BATCH, 256, 0, stream>>>(dec_i, E_ph, xin);
        gemm(xin,   2 * DM, wDWih, dec_bih, gid, G3, nullptr, 0, BATCH, G3, 2 * DM, 0);
        gemm(hdecb, DM,     wDWhh, dec_bhh, ghd, G3, nullptr, 0, BATCH, G3, DM, 0);
        gru_pointwise_k<<<BATCH, 256, 0, stream>>>(gid, ghd, hdec, hdecb, DM, nullptr, 0);
        gemm(hdecb, DM, wPost, post_b, out_res + (size_t)ti * D_PH, SEQ_LP * D_PH,
             nullptr, 0, BATCH, D_PH, DM, 0);
        if (ti < SEQ_LP - 1) {
            attend_k<<<BATCH, 128, 0, stream>>>(
                hdec, kbuf, vbuf, out_attn + (size_t)(ti + 1) * SEQ_LT, xin + DM);
            argmax_k<<<1, 256, 0, stream>>>(
                out_res + (size_t)ti * D_PH, SEQ_LP * D_PH, dec_i);
        }
    }
}
